// Involution2d_80083960201288
// MI455X (gfx1250) — compile-verified, matrix operand order checked by reading, not run on hardware
//
#include <hip/hip_runtime.h>

// Problem constants (match reference)
#define B_   4
#define C_   256
#define H_   64
#define W_   64
#define G_   16
#define K_   7
#define P_   3
#define RC_  64           // C / R
#define KKG_ 784          // K*K*G

// Fragment-ready workspace layout (all bf16):
//   xf : [b][h][wt][ks(8)][lane(32)][j(16)]   -> 4*64*4*8*32*16 = 4,194,304
//   wrf: [mt(4)][ks(8)][lane(32)][j(16)]      -> 16,384
//   wsf: [mt(49)][ks(2)][lane(32)][j(16)]     -> 50,176
#define XF_ELEMS  (B_ * H_ * 4 * 8 * 32 * 16)
#define WRF_ELEMS (4 * 8 * 32 * 16)
#define WSF_ELEMS (49 * 2 * 32 * 16)

typedef __attribute__((ext_vector_type(16))) __bf16 v16bf;
typedef __attribute__((ext_vector_type(8)))  float  v8f;

// ISA 7.12.2: 16-bit A/B fragment K index for (lane, slot j)
__device__ __forceinline__ int kmap(int lane, int j) {
  const int hi = (lane >> 4) & 1;
  return (j < 8) ? ((hi ? 8 : 0) + j) : ((hi ? 24 : 16) + (j - 8));
}

// ---- Pre-pass 1: weights f32 -> bf16, fragment-ready swizzle ----
__global__ __launch_bounds__(256) void convert_w_frag(
    const float* __restrict__ wr, const float* __restrict__ wsp,
    __bf16* __restrict__ wrf, __bf16* __restrict__ wsf) {
  const int i = blockIdx.x * 256 + threadIdx.x;
  if (i < WRF_ELEMS) {
    const int j = i & 15, lane = (i >> 4) & 31, ks = (i >> 9) & 7, mt = i >> 12;
    const int row = mt * 16 + (lane & 15);
    const int k = ks * 32 + kmap(lane, j);
    wrf[i] = (__bf16)wr[row * C_ + k];
  }
  if (i < WSF_ELEMS) {
    const int j = i & 15, lane = (i >> 4) & 31, ks = (i >> 9) & 1, mt = i >> 10;
    const int row = mt * 16 + (lane & 15);
    const int k = ks * 32 + kmap(lane, j);
    wsf[i] = (__bf16)wsp[row * RC_ + k];
  }
}

// ---- Pre-pass 2: x f32 -> bf16, fragment-ready swizzle (8.4 MB, one-time) ----
__global__ __launch_bounds__(256) void convert_x_frag(
    const float* __restrict__ x, __bf16* __restrict__ xf) {
  const int t = blockIdx.x * 256 + threadIdx.x;   // XF_ELEMS threads
  const int j    = t & 15;
  const int lane = (t >> 4) & 31;
  const int ks   = (t >> 9) & 7;
  const int wt   = (t >> 12) & 3;
  const int h    = (t >> 14) & 63;
  const int b    = t >> 20;
  const int col  = lane & 15;
  const int k    = ks * 32 + kmap(lane, j);
  xf[t] = (__bf16)x[(((size_t)b * C_ + k) * H_ + h) * W_ + wt * 16 + col];
}

// ---- Fused main kernel: fragments load as contiguous b128, no kmap math ----
__global__ __launch_bounds__(128) void involution_fast(
    const float* __restrict__ x, const float* __restrict__ b_span,
    const __bf16* __restrict__ xf, const __bf16* __restrict__ wrf,
    const __bf16* __restrict__ wsf, float* __restrict__ out) {
  __shared__ __bf16 lds_red[2 * 32 * 16];              //  2 KB fragment-ready red
  __shared__ alignas(16) float lds_ker[KKG_ * 16];     // 50 KB per-pixel kernels

  const int tid  = threadIdx.x;
  const int lane = tid & 31;
  const int wave = tid >> 5;
  const int col  = lane & 15;
  const int hi8  = (lane >> 4) ? 8 : 0;

  int bx = blockIdx.x;
  const int wt = bx & 3;  const int w0 = wt * 16;  bx >>= 2;
  const int h  = bx & 63;
  const int b  = bx >> 6;

  // ---- Stage 1: red(64x16) = w_reduce @ x_tile; dual accumulators ----
  {
    const __bf16* xfp = xf + (size_t)(((b * H_ + h) * 4 + wt)) * (8 * 32 * 16);
    const __bf16* wrp = wrf + (size_t)wave * (8 * 32 * 16);
    v8f accA = {}, accB = {};
#pragma unroll
    for (int ks = 0; ks < 8; ++ks) {
      const v16bf a   = *(const v16bf*)(wrp + (ks * 32 + lane) * 16);
      const v16bf bbv = *(const v16bf*)(xfp + (ks * 32 + lane) * 16);
      if (ks & 1)
        accB = __builtin_amdgcn_wmma_f32_16x16x32_bf16(false, a, false, bbv,
                                                       (short)0, accB, false, false);
      else
        accA = __builtin_amdgcn_wmma_f32_16x16x32_bf16(false, a, false, bbv,
                                                       (short)0, accA, false, false);
    }
    const v8f acc = accA + accB;
    // Scatter red into fragment-ready LDS layout. For this thread the 8
    // destinations are contiguous (one ds_store_b128):
    //   m = wave*16 + hi8 + i ; kk = m&31 is an aligned block of 8
    const int m0  = wave * 16 + hi8;
    const int kk0 = m0 & 31;
    const int ks2 = m0 >> 5;
    const int hi2 = (kk0 >> 3) & 1;
    const int j20 = (kk0 < 16) ? (kk0 & 7) : (8 + (kk0 & 7));
    __bf16* dst = lds_red + (ks2 * 32 + hi2 * 16 + col) * 16 + j20;
#pragma unroll
    for (int i = 0; i < 8; ++i) dst[i] = (__bf16)acc[i];
  }
  __syncthreads();

  // ---- Stage 2: ker(784x16) = w_span @ red + b_span ----
  {
    // B fragments identical for all M tiles: two contiguous 32B LDS loads each.
    const v16bf bb0 = *(const v16bf*)(lds_red + (0 * 32 + lane) * 16);
    const v16bf bb1 = *(const v16bf*)(lds_red + (1 * 32 + lane) * 16);
    for (int mt = wave; mt < 49; mt += 4) {            // wave-uniform trip count
      const int mbase = mt * 16 + hi8;
      v8f acc0, acc1 = {};
#pragma unroll
      for (int i = 0; i < 8; ++i) acc0[i] = b_span[mbase + i];
      const __bf16* wp = wsf + (size_t)mt * (2 * 32 * 16);
      const v16bf a0 = *(const v16bf*)(wp + lane * 16);
      const v16bf a1 = *(const v16bf*)(wp + (32 + lane) * 16);
      acc0 = __builtin_amdgcn_wmma_f32_16x16x32_bf16(false, a0, false, bb0,
                                                     (short)0, acc0, false, false);
      acc1 = __builtin_amdgcn_wmma_f32_16x16x32_bf16(false, a1, false, bb1,
                                                     (short)0, acc1, false, false);
#pragma unroll
      for (int i = 0; i < 8; ++i)
        lds_ker[(mbase + i) * 16 + col] = acc0[i] + acc1[i];
    }
  }
  __syncthreads();

  // ---- Stage 3: out[c,p] = sum_{dy,dx} ker[g, dy*7+dx, p] * x[c, h+dy, w0+p+dx] ----
  for (int cpass = 0; cpass < 2; ++cpass) {
    const int c = tid + cpass * 128;
    const int g = c >> 4;
    const float* xc = x + (size_t)(b * C_ + c) * (H_ * W_);
    float acc[16];
#pragma unroll
    for (int p = 0; p < 16; ++p) acc[p] = 0.0f;

    for (int dy = -P_; dy <= P_; ++dy) {
      const int ry = h + dy;
      float xv[16 + K_ - 1];
      if (ry >= 0 && ry < H_) {
        const float* xr = xc + (size_t)ry * W_;
#pragma unroll
        for (int i = 0; i < 22; ++i) {
          const int cx = w0 - P_ + i;
          xv[i] = (cx >= 0 && cx < W_) ? xr[cx] : 0.0f;
        }
      } else {
#pragma unroll
        for (int i = 0; i < 22; ++i) xv[i] = 0.0f;
      }
      const int kb = (g * 49 + (dy + P_) * K_) * 16;
#pragma unroll
      for (int dx = 0; dx < K_; ++dx) {
        const float4* kr = (const float4*)&lds_ker[kb + dx * 16];  // 64B aligned
        const float4 k0 = kr[0], k1 = kr[1], k2 = kr[2], k3 = kr[3];
        const float* kw = (const float*)&k0;  (void)kw;
        const float kvv[16] = {k0.x, k0.y, k0.z, k0.w, k1.x, k1.y, k1.z, k1.w,
                               k2.x, k2.y, k2.z, k2.w, k3.x, k3.y, k3.z, k3.w};
#pragma unroll
        for (int p = 0; p < 16; ++p)
          acc[p] += kvv[p] * xv[p + dx];
      }
    }
    float* op = out + (size_t)(b * C_ + c) * (H_ * W_) + (size_t)h * W_ + w0;
#pragma unroll
    for (int p = 0; p < 16; ++p) op[p] = acc[p];
  }
}

// ---- Fallback: self-contained (inline bf16 convert + LDS staging) ----
__global__ __launch_bounds__(128) void involution_generic(
    const float* __restrict__ x, const float* __restrict__ w_reduce,
    const float* __restrict__ w_span, const float* __restrict__ b_span,
    float* __restrict__ out) {
  __shared__ __bf16 lds_xbf[C_ * 16];
  __shared__ __bf16 lds_red[RC_ * 16];
  __shared__ alignas(16) float lds_ker[KKG_ * 16];

  const int tid  = threadIdx.x;
  const int lane = tid & 31;
  const int wave = tid >> 5;
  const int col  = lane & 15;
  const int hi8  = (lane >> 4) ? 8 : 0;

  int bx = blockIdx.x;
  const int w0 = (bx & 3) * 16;  bx >>= 2;
  const int h  = bx & 63;
  const int b  = bx >> 6;

  const float* xp = x + ((size_t)b * C_) * (H_ * W_) + (size_t)h * W_ + w0;
  for (int k = tid; k < C_; k += 128) {
    const float* r = xp + (size_t)k * (H_ * W_);
#pragma unroll
    for (int p = 0; p < 16; ++p) lds_xbf[k * 16 + p] = (__bf16)r[p];
  }
  __syncthreads();

  {
    v8f acc = {};
    const int row = wave * 16 + col;
#pragma unroll
    for (int ks = 0; ks < 8; ++ks) {
      const int k0 = ks * 32;
      v16bf a, bbv;
#pragma unroll
      for (int j = 0; j < 16; ++j) {
        const int k = k0 + kmap(lane, j);
        a[j]   = (__bf16)w_reduce[row * C_ + k];
        bbv[j] = lds_xbf[k * 16 + col];
      }
      acc = __builtin_amdgcn_wmma_f32_16x16x32_bf16(false, a, false, bbv,
                                                    (short)0, acc, false, false);
    }
    const int mbase = wave * 16 + hi8;
#pragma unroll
    for (int i = 0; i < 8; ++i)
      lds_red[(mbase + i) * 16 + col] = (__bf16)acc[i];
  }
  __syncthreads();

  {
    v16bf bb0, bb1;
#pragma unroll
    for (int j = 0; j < 16; ++j) {
      const int ka = kmap(lane, j);
      bb0[j] = lds_red[ka * 16 + col];
      bb1[j] = lds_red[(32 + ka) * 16 + col];
    }
    for (int mt = wave; mt < 49; mt += 4) {
      const int mbase = mt * 16 + hi8;
      v8f acc0, acc1 = {};
#pragma unroll
      for (int i = 0; i < 8; ++i) acc0[i] = b_span[mbase + i];
      const int row = mt * 16 + col;
      v16bf a0, a1;
#pragma unroll
      for (int j = 0; j < 16; ++j) {
        const int ka = kmap(lane, j);
        a0[j] = (__bf16)w_span[row * RC_ + ka];
        a1[j] = (__bf16)w_span[row * RC_ + 32 + ka];
      }
      acc0 = __builtin_amdgcn_wmma_f32_16x16x32_bf16(false, a0, false, bb0,
                                                     (short)0, acc0, false, false);
      acc1 = __builtin_amdgcn_wmma_f32_16x16x32_bf16(false, a1, false, bb1,
                                                     (short)0, acc1, false, false);
#pragma unroll
      for (int i = 0; i < 8; ++i)
        lds_ker[(mbase + i) * 16 + col] = acc0[i] + acc1[i];
    }
  }
  __syncthreads();

  for (int cpass = 0; cpass < 2; ++cpass) {
    const int c = tid + cpass * 128;
    const int g = c >> 4;
    const float* xc = x + (size_t)(b * C_ + c) * (H_ * W_);
    float acc[16];
#pragma unroll
    for (int p = 0; p < 16; ++p) acc[p] = 0.0f;
    for (int dy = -P_; dy <= P_; ++dy) {
      const int ry = h + dy;
      float xv[16 + K_ - 1];
      if (ry >= 0 && ry < H_) {
        const float* xr = xc + (size_t)ry * W_;
#pragma unroll
        for (int i = 0; i < 22; ++i) {
          const int cx = w0 - P_ + i;
          xv[i] = (cx >= 0 && cx < W_) ? xr[cx] : 0.0f;
        }
      } else {
#pragma unroll
        for (int i = 0; i < 22; ++i) xv[i] = 0.0f;
      }
      const int kb = (g * 49 + (dy + P_) * K_) * 16;
#pragma unroll
      for (int dx = 0; dx < K_; ++dx) {
#pragma unroll
        for (int p = 0; p < 16; ++p)
          acc[p] += lds_ker[kb + dx * 16 + p] * xv[p + dx];
      }
    }
    float* op = out + (size_t)(b * C_ + c) * (H_ * W_) + (size_t)h * W_ + w0;
#pragma unroll
    for (int p = 0; p < 16; ++p) op[p] = acc[p];
  }
}

extern "C" void kernel_launch(void* const* d_in, const int* in_sizes, int n_in,
                              void* d_out, int out_size, void* d_ws, size_t ws_size,
                              hipStream_t stream) {
  (void)in_sizes; (void)n_in; (void)out_size;
  const float* x        = (const float*)d_in[0];
  const float* w_reduce = (const float*)d_in[1];
  const float* w_span   = (const float*)d_in[2];
  const float* b_span   = (const float*)d_in[3];
  float* out            = (float*)d_out;

  dim3 grid((W_ / 16) * H_ * B_);  // 1024 workgroups, 16-pixel tiles
  dim3 block(128);                 // 4 wave32

  const size_t need = (size_t)(XF_ELEMS + WRF_ELEMS + WSF_ELEMS) * sizeof(__bf16);
  if (ws_size >= need && d_ws != nullptr) {
    __bf16* xf  = (__bf16*)d_ws;
    __bf16* wrf = xf + XF_ELEMS;
    __bf16* wsf = wrf + WRF_ELEMS;
    convert_w_frag<<<(WSF_ELEMS + 255) / 256, 256, 0, stream>>>(w_reduce, w_span,
                                                                wrf, wsf);
    convert_x_frag<<<XF_ELEMS / 256, 256, 0, stream>>>(x, xf);
    involution_fast<<<grid, block, 0, stream>>>(x, b_span, xf, wrf, wsf, out);
  } else {
    involution_generic<<<grid, block, 0, stream>>>(x, w_reduce, w_span, b_span, out);
  }
}
